// LSTMencoder_8873402434188
// MI455X (gfx1250) — matrix operand order repaired and checked by reading, not compiled, not run on hardware
//
#include <hip/hip_runtime.h>
#include <hip/hip_bf16.h>

typedef __bf16 bf16_t;
typedef __attribute__((ext_vector_type(16))) __bf16 v16bf;
typedef __attribute__((ext_vector_type(8)))  float  v8f;

#define T_LEN 2048
#define I_DIM 1024
#define H_DIM 2048
#define G4    (4 * H_DIM)   // 8192 gate rows

// ---- workspace layout (bytes) ----
#define XG_OFF  0                                   // x_gates fp32: T x 4H = 64 MB
#define XB_OFF  (XG_OFF + (size_t)T_LEN * G4 * 4)   // x bf16:  T x I  = 4 MB
#define WB_OFF  (XB_OFF + (size_t)T_LEN * I_DIM * 2)// w_ih bf16: 4H x I = 16 MB
#define H_OFF   (WB_OFF + (size_t)G4 * I_DIM * 2)   // h double buffer: 2 x H fp32
#define BAR_OFF (H_OFF + (size_t)2 * H_DIM * 4)     // barrier: 2 ints

// ---- recurrence partitioning ----
#define NB   128              // persistent blocks (must be co-resident)
#define BT   256              // threads per block = 8 waves
#define JPB  (H_DIM / NB)     // 16 hidden units per block
#define RPB  (4 * JPB)        // 64 gate rows per block
#define RPW  (RPB / 8)        // 8 rows per wave

// ------------------------------------------------------------------
// init: zero barrier, copy h0 into h double-buffer slot 0
// ------------------------------------------------------------------
__global__ void init_kernel(const float* __restrict__ h0,
                            float* __restrict__ ws_h,
                            int* __restrict__ bar) {
  int tid = threadIdx.x;
  if (tid < 2) bar[tid] = 0;
  for (int j = tid; j < H_DIM; j += blockDim.x) ws_h[j] = h0[j];
}

// ------------------------------------------------------------------
// fp32 -> bf16 conversion (grid-stride)
// ------------------------------------------------------------------
__global__ void cvt_f32_bf16(const float* __restrict__ src,
                             bf16_t* __restrict__ dst, int n) {
  int i = blockIdx.x * blockDim.x + threadIdx.x;
  int stride = gridDim.x * blockDim.x;
  for (; i < n; i += stride) dst[i] = (bf16_t)src[i];
}

// ------------------------------------------------------------------
// Phase 1: x_gates[t][m] = sum_k x[t][k] * w_ih[m][k]   (bf16 WMMA, f32 acc)
// One 16x16 C tile per wave; 8 waves/block share the same A rows (L1 reuse).
// Fragment layouts per CDNA5 ISA 7.12.2 (16-bit A 16x32, B 32x16, f32 C).
// ------------------------------------------------------------------
__global__ void __launch_bounds__(256)
gemm_xgates(const bf16_t* __restrict__ xb,
            const bf16_t* __restrict__ wb,
            float* __restrict__ xg) {
  const int wave = threadIdx.x >> 5;
  const int lane = threadIdx.x & 31;
  const int half = lane >> 4;     // 0: lanes 0-15, 1: lanes 16-31
  const int lr   = lane & 15;

  const int TM   = G4 / 16;       // 512 tiles along gate dim
  const int tile = blockIdx.x * 8 + wave;
  const int t0   = (tile / TM) * 16;
  const int m0   = (tile % TM) * 16;

  const bf16_t* arow = xb + (size_t)(t0 + lr) * I_DIM;  // A: row = M (=t), striped by K
  const bf16_t* brow = wb + (size_t)(m0 + lr) * I_DIM;  // B: col = N (=m), striped by K

  v8f acc = {};
  for (int k0 = 0; k0 < I_DIM; k0 += 32) {
    v16bf a, b;
#pragma unroll
    for (int i = 0; i < 8; ++i) {
      // A 16x32 bf16: lo lanes hold K={0..7,16..23}, hi lanes K={8..15,24..31}
      int ka = ((i & 4) ? 16 : 0) + half * 8 + 2 * (i & 3);
      a[2 * i]     = arow[k0 + ka];
      a[2 * i + 1] = arow[k0 + ka + 1];
      // B 32x16 bf16: lo lanes K={0..7,16..23}, hi lanes K={8..15,24..31} (2 K per VGPR)
      int kb = ((i & 4) ? 16 : 0) + half * 8 + 2 * (i & 3);
      b[2 * i]     = brow[k0 + kb];
      b[2 * i + 1] = brow[k0 + kb + 1];
    }
    acc = __builtin_amdgcn_wmma_f32_16x16x32_bf16(
        false, a, false, b, (short)0, acc, false, false);
  }
  // C layout: VGPR r -> row M = r + 8*half, col N = lr
#pragma unroll
  for (int r = 0; r < 8; ++r)
    xg[(size_t)(t0 + half * 8 + r) * G4 + m0 + lr] = acc[r];
}

// ------------------------------------------------------------------
// device-wide split barrier (persistent-kernel sync, agent scope)
// ------------------------------------------------------------------
__device__ __forceinline__ void grid_barrier(int* bar, int nb) {
  __syncthreads();
  if (threadIdx.x == 0) {
    int g = __hip_atomic_load(&bar[1], __ATOMIC_RELAXED, __HIP_MEMORY_SCOPE_AGENT);
    int prev = __hip_atomic_fetch_add(&bar[0], 1, __ATOMIC_ACQ_REL,
                                      __HIP_MEMORY_SCOPE_AGENT);
    if (prev == nb - 1) {
      __hip_atomic_store(&bar[0], 0, __ATOMIC_RELAXED, __HIP_MEMORY_SCOPE_AGENT);
      __hip_atomic_fetch_add(&bar[1], 1, __ATOMIC_ACQ_REL,
                             __HIP_MEMORY_SCOPE_AGENT);
    } else {
      while (__hip_atomic_load(&bar[1], __ATOMIC_ACQUIRE,
                               __HIP_MEMORY_SCOPE_AGENT) == g) {
        __builtin_amdgcn_s_sleep(1);
      }
    }
  }
  __syncthreads();
}

__device__ __forceinline__ float sigmoid_f(float x) {
  return 1.0f / (1.0f + __expf(-x));
}
__device__ __forceinline__ float tanh_f(float x) {
  float e = __expf(2.0f * x);           // inf-safe: x>>0 -> 1, x<<0 -> -1
  return 1.0f - 2.0f / (e + 1.0f);
}

// ------------------------------------------------------------------
// Phase 2: persistent recurrence. Block b owns hidden units
// [b*JPB, (b+1)*JPB) i.e. gate rows {g*H + j} for g=0..3. c stays in
// registers; h double-buffered in ws; one grid barrier per timestep.
// ------------------------------------------------------------------
__global__ void __launch_bounds__(BT)
lstm_recurrence(const float* __restrict__ w_hh,
                const float* __restrict__ xg,
                const float* __restrict__ b_ih,
                const float* __restrict__ b_hh,
                const float* __restrict__ c0,
                float* __restrict__ ws_h,   // 2 buffers of H
                int* __restrict__ bar,
                float* __restrict__ out) {
  __shared__ float hsh[H_DIM];
  __shared__ float gsh[RPB];

  const int tid  = threadIdx.x;
  const int wave = tid >> 5;
  const int lane = tid & 31;
  const int b    = blockIdx.x;
  const int j0   = b * JPB;

  float c_reg = 0.0f;
  if (tid < JPB) c_reg = c0[j0 + tid];

  for (int t = 0; t < T_LEN; ++t) {
    float* hr = ws_h + (t & 1) * H_DIM;         // read h_{t-1}
    float* hw = ws_h + ((t + 1) & 1) * H_DIM;   // write h_t

    // stage h into LDS (agent-scope loads: cross-WGP coherence)
    for (int j = tid; j < H_DIM; j += BT) {
      int vi = __hip_atomic_load((int*)&hr[j], __ATOMIC_RELAXED,
                                 __HIP_MEMORY_SCOPE_AGENT);
      hsh[j] = __int_as_float(vi);
    }
    __syncthreads();

    // each wave computes RPW gate rows: full-length fp32 dot from L2 + LDS
#pragma unroll
    for (int r = 0; r < RPW; ++r) {
      int q    = wave * RPW + r;
      int gate = q >> 4;            // q / JPB
      int jj   = q & (JPB - 1);
      int row  = gate * H_DIM + j0 + jj;
      const float4* wrow = (const float4*)(w_hh + (size_t)row * H_DIM);
      const float4* h4   = (const float4*)hsh;
      float4 a4 = {0.f, 0.f, 0.f, 0.f};
      for (int k = lane; k < H_DIM / 4; k += 32) {
        float4 wv = wrow[k];
        float4 hv = h4[k];
        a4.x += wv.x * hv.x; a4.y += wv.y * hv.y;
        a4.z += wv.z * hv.z; a4.w += wv.w * hv.w;
      }
      float s = (a4.x + a4.y) + (a4.z + a4.w);
#pragma unroll
      for (int off = 16; off > 0; off >>= 1) s += __shfl_xor(s, off, 32);
      if (lane == 0)
        gsh[q] = s + xg[(size_t)t * G4 + row] + b_ih[row] + b_hh[row];
    }
    __syncthreads();

    // cell update for this block's JPB hidden units (gate order i,f,g,o)
    if (tid < JPB) {
      float iv = sigmoid_f(gsh[tid]);
      float fv = sigmoid_f(gsh[JPB + tid]);
      float gv = tanh_f(gsh[2 * JPB + tid]);
      float ov = sigmoid_f(gsh[3 * JPB + tid]);
      c_reg = fv * c_reg + iv * gv;
      float hv = ov * tanh_f(c_reg);
      out[(size_t)t * H_DIM + j0 + tid] = hv;
      __hip_atomic_store((int*)&hw[j0 + tid], __float_as_int(hv),
                         __ATOMIC_RELAXED, __HIP_MEMORY_SCOPE_AGENT);
    }
    grid_barrier(bar, NB);
  }
}

// ------------------------------------------------------------------
extern "C" void kernel_launch(void* const* d_in, const int* in_sizes, int n_in,
                              void* d_out, int out_size, void* d_ws, size_t ws_size,
                              hipStream_t stream) {
  const float* x    = (const float*)d_in[0];
  const float* h0   = (const float*)d_in[1];
  const float* c0   = (const float*)d_in[2];
  const float* w_ih = (const float*)d_in[3];
  const float* w_hh = (const float*)d_in[4];
  const float* b_ih = (const float*)d_in[5];
  const float* b_hh = (const float*)d_in[6];

  char*   ws   = (char*)d_ws;
  float*  xg   = (float*)(ws + XG_OFF);
  bf16_t* xb   = (bf16_t*)(ws + XB_OFF);
  bf16_t* wb   = (bf16_t*)(ws + WB_OFF);
  float*  ws_h = (float*)(ws + H_OFF);
  int*    bar  = (int*)(ws + BAR_OFF);
  float*  out  = (float*)d_out;

  init_kernel<<<1, 256, 0, stream>>>(h0, ws_h, bar);
  cvt_f32_bf16<<<1024, 256, 0, stream>>>(x, xb, T_LEN * I_DIM);
  cvt_f32_bf16<<<4096, 256, 0, stream>>>(w_ih, wb, G4 * I_DIM);

  // (T/16)*(4H/16) = 65536 C tiles, 8 waves (tiles) per block
  gemm_xgates<<<8192, 256, 0, stream>>>(xb, wb, xg);

  lstm_recurrence<<<NB, BT, 0, stream>>>(w_hh, xg, b_ih, b_hh, c0,
                                         ws_h, bar, out);
}
